// ConvPersonGraphical_83631603187842
// MI455X (gfx1250) — compile-verified, matrix operand order checked by reading, not run on hardware
//
#include <hip/hip_runtime.h>

typedef __bf16 bf16_t;
typedef __attribute__((ext_vector_type(16))) __bf16 bf16x16;
typedef __attribute__((ext_vector_type(8)))  __bf16 bf16x8;
typedef __attribute__((ext_vector_type(8)))  float  floatx8;

#define N_   64
#define V_   512
#define C_   512
#define K_   2
#define OUT_ 512
#define O2_  1024   // 2*OUT

// ---------------------------------------------------------------------------
// f32 -> bf16 cast (for x and conv_w)
// ---------------------------------------------------------------------------
__global__ void cast_f32_to_bf16(const float* __restrict__ src,
                                 bf16_t* __restrict__ dst, long n) {
    long i = (long)blockIdx.x * blockDim.x + threadIdx.x;
    if (i < n) dst[i] = (bf16_t)src[i];
}

// ---------------------------------------------------------------------------
// A prep: per (n,k) 32x32 tile transpose.
//   At_f32[n,k,e,m] = A[n,k,m,e]        (reference output #2, f32 -> d_out)
//   Abt   [n,k,e,m] = bf16(A[n,k,m,e])  (B^T for GEMM2: rows e, K-dim m)
//   Ar    [n,m,k*V+e] = bf16(A[n,k,m,e]) (B^T for GEMM3: rows m, fused K)
// ---------------------------------------------------------------------------
__global__ void prep_A(const float* __restrict__ A,
                       float* __restrict__ At_f32,
                       bf16_t* __restrict__ Abt,
                       bf16_t* __restrict__ Ar) {
    __shared__ float tile[32][33];
    int b = blockIdx.z;            // b = n*K + k
    int n = b >> 1, k = b & 1;
    int e0 = blockIdx.x * 32, m0 = blockIdx.y * 32;
    const float* Ab = A + (long)b * V_ * V_;
#pragma unroll
    for (int j = 0; j < 4; ++j) {
        int m = m0 + threadIdx.y + j * 8;
        int e = e0 + threadIdx.x;
        float v = Ab[(long)m * V_ + e];
        tile[threadIdx.y + j * 8][threadIdx.x] = v;
        Ar[((long)n * V_ + m) * (K_ * V_) + k * V_ + e] = (bf16_t)v;
    }
    __syncthreads();
#pragma unroll
    for (int j = 0; j < 4; ++j) {
        int e = e0 + threadIdx.y + j * 8;
        int m = m0 + threadIdx.x;
        float v = tile[threadIdx.x][threadIdx.y + j * 8];
        long idx = ((long)b * V_ + e) * V_ + m;
        At_f32[idx] = v;
        Abt[idx]    = (bf16_t)v;
    }
}

// ---------------------------------------------------------------------------
// WMMA fragment load, CDNA5 16-bit A/B layout (CDNA5 ISA §7.12.2):
//   lanes 0-15 : M=lane,    VGPR0-3 K=k0+0..7,  VGPR4-7 K=k0+16..23
//   lanes 16-31: M=lane-16, VGPR0-3 K=k0+8..15, VGPR4-7 K=k0+24..31
// Two contiguous 128-bit loads per lane.
// ---------------------------------------------------------------------------
__device__ __forceinline__ bf16x16 load_frag(const bf16_t* __restrict__ base,
                                             int ld, int row, int k0, int lane) {
    const bf16_t* p = base + (long)(row + (lane & 15)) * ld + k0 + ((lane >> 4) << 3);
    bf16x8 lo = *(const bf16x8*)p;
    bf16x8 hi = *(const bf16x8*)(p + 16);
    return __builtin_shufflevector(lo, hi, 0,1,2,3,4,5,6,7,8,9,10,11,12,13,14,15);
}

__device__ __forceinline__ void store_tile(floatx8 acc, long cbase, int m0, int n0,
                                           int ldc, const float* __restrict__ bias,
                                           float* __restrict__ Cf,
                                           bf16_t* __restrict__ Cb) {
#pragma unroll
    for (int r = 0; r < 8; ++r) {
        int m = m0 + r;
        float val = acc[r] + (bias ? bias[m] : 0.0f);
        long idx = cbase + (long)m * ldc + n0;
        if (Cf) Cf[idx] = val; else Cb[idx] = (bf16_t)val;
    }
}

// ---------------------------------------------------------------------------
// Generic batched bf16 GEMM, f32 accumulate:  C[b] = A[b] (MxK) * Bt[b]^T (KxN)
// A row-major (lda=Kdim), Bt row-major NxK (ldb=Kdim).
// Each wave: 32(M) x 64(N) tile -> 2 A-frags + 4 B-frags, 8 WMMAs per K-step.
// Block = 8 waves covering 256(M) x 64(N).
// C batch offset = (b/batchLo)*cStrideHi + (b%batchLo)*cStrideLo.
// ---------------------------------------------------------------------------
__global__ void gemm_bf16(const bf16_t* __restrict__ Amat, long aStride,
                          const bf16_t* __restrict__ Bt,   long bStride,
                          const float* __restrict__ bias,
                          float* __restrict__ Cf, bf16_t* __restrict__ Cb,
                          int M, int Kdim, int ldc,
                          int batchLo, long cStrideHi, long cStrideLo) {
    int tid  = threadIdx.x;
    int wave = tid >> 5, lane = tid & 31;
    int b = blockIdx.z;
    int tileM = blockIdx.x * 256 + wave * 32;
    int tileN = blockIdx.y * 64;
    if (tileM >= M) return;   // wave-uniform; EXEC stays all-1s for WMMA

    const bf16_t* Ab = Amat + (long)b * aStride;
    const bf16_t* Bb = Bt   + (long)b * bStride;

    floatx8 acc[2][4];
#pragma unroll
    for (int i = 0; i < 2; ++i)
#pragma unroll
        for (int j = 0; j < 4; ++j) acc[i][j] = (floatx8){};

    for (int k0 = 0; k0 < Kdim; k0 += 32) {
        bf16x16 af[2];
        af[0] = load_frag(Ab, Kdim, tileM +  0, k0, lane);
        af[1] = load_frag(Ab, Kdim, tileM + 16, k0, lane);
        bf16x16 bfv[4];
#pragma unroll
        for (int j = 0; j < 4; ++j)
            bfv[j] = load_frag(Bb, Kdim, tileN + j * 16, k0, lane);
#pragma unroll
        for (int i = 0; i < 2; ++i)
#pragma unroll
            for (int j = 0; j < 4; ++j)
                acc[i][j] = __builtin_amdgcn_wmma_f32_16x16x32_bf16(
                    false, af[i], false, bfv[j], (short)0, acc[i][j], false, false);
    }

    long cbase = (long)(b / batchLo) * cStrideHi + (long)(b % batchLo) * cStrideLo;
    int  rowOff = (lane < 16) ? 0 : 8;   // C/D layout: VGPR r -> M = r (+8 hi lanes)
    int  col    = lane & 15;
#pragma unroll
    for (int i = 0; i < 2; ++i)
#pragma unroll
        for (int j = 0; j < 4; ++j)
            store_tile(acc[i][j], cbase, tileM + i * 16 + rowOff,
                       tileN + j * 16 + col, ldc, bias, Cf, Cb);
}

// ---------------------------------------------------------------------------
// BatchNorm stats: one block per channel o, deterministic LDS reduction over
// the 32768 (n,v) samples. Emits fused scale/shift.
// ---------------------------------------------------------------------------
__global__ void bn_stats(const bf16_t* __restrict__ h,
                         const float* __restrict__ gamma,
                         const float* __restrict__ beta,
                         float* __restrict__ scale, float* __restrict__ shift) {
    __shared__ float ssum[256], ssq[256];
    int o = blockIdx.x;
    float s = 0.f, sq = 0.f;
    for (int idx = threadIdx.x; idx < N_ * V_; idx += 256) {
        int n = idx >> 9, v = idx & (V_ - 1);
        float val = (float)h[((long)n * O2_ + o) * V_ + v];
        s += val; sq += val * val;
    }
    ssum[threadIdx.x] = s; ssq[threadIdx.x] = sq;
    __syncthreads();
    for (int st = 128; st > 0; st >>= 1) {
        if (threadIdx.x < st) {
            ssum[threadIdx.x] += ssum[threadIdx.x + st];
            ssq[threadIdx.x]  += ssq[threadIdx.x + st];
        }
        __syncthreads();
    }
    if (threadIdx.x == 0) {
        const float inv = 1.0f / (float)(N_ * V_);
        float mean = ssum[0] * inv;
        float var  = ssq[0] * inv - mean * mean;
        float sc   = rsqrtf(var + 1e-5f) * gamma[o];
        scale[o] = sc;
        shift[o] = beta[o] - mean * sc;
    }
}

__global__ void bn_apply(bf16_t* __restrict__ h,
                         const float* __restrict__ scale,
                         const float* __restrict__ shift) {
    long i = (long)blockIdx.x * blockDim.x + threadIdx.x;  // grid sized exactly
    int o = (int)((i >> 9) & (O2_ - 1));
    h[i] = (bf16_t)((float)h[i] * scale[o] + shift[o]);
}

// ---------------------------------------------------------------------------
// out[n,v,c] = (1+eps)*x[n,v,c] + y[n,c,v]   (LDS 32x32 transpose of y)
// ---------------------------------------------------------------------------
__global__ void epilogue(const float* __restrict__ x, const float* __restrict__ y,
                         const float* __restrict__ eps_p, float* __restrict__ out) {
    __shared__ float tile[32][33];
    int n = blockIdx.z;
    int v0 = blockIdx.x * 32, c0 = blockIdx.y * 32;
    const float* yb = y + (long)n * OUT_ * V_;
#pragma unroll
    for (int j = 0; j < 4; ++j) {
        int c = c0 + threadIdx.y + j * 8;
        int v = v0 + threadIdx.x;
        tile[threadIdx.y + j * 8][threadIdx.x] = yb[(long)c * V_ + v];
    }
    __syncthreads();
    float e = 1.0f + eps_p[0];
#pragma unroll
    for (int j = 0; j < 4; ++j) {
        int v = v0 + threadIdx.y + j * 8;
        int c = c0 + threadIdx.x;
        long idx = ((long)n * V_ + v) * C_ + c;
        out[idx] = e * x[idx] + tile[threadIdx.x][threadIdx.y + j * 8];
    }
}

// ---------------------------------------------------------------------------
extern "C" void kernel_launch(void* const* d_in, const int* in_sizes, int n_in,
                              void* d_out, int out_size, void* d_ws, size_t ws_size,
                              hipStream_t stream) {
    const float* x      = (const float*)d_in[0];
    const float* A      = (const float*)d_in[1];
    const float* conv_w = (const float*)d_in[2];
    const float* conv_b = (const float*)d_in[3];
    const float* gamma  = (const float*)d_in[4];
    const float* beta   = (const float*)d_in[5];
    const float* eps_p  = (const float*)d_in[6];

    float* out    = (float*)d_out;
    float* At_out = out + (long)N_ * V_ * C_;   // output #2: A^T (f32)

    // workspace carve-out (~370 MB)
    char* ws = (char*)d_ws;
    size_t off = 0;
    auto carve = [&](size_t bytes) -> void* {
        void* p = ws + off;
        off = (off + bytes + 255) & ~(size_t)255;
        return p;
    };
    bf16_t* xb   = (bf16_t*)carve((size_t)N_ * V_ * C_ * 2);
    bf16_t* wb   = (bf16_t*)carve((size_t)O2_ * C_ * 2);
    bf16_t* Abt  = (bf16_t*)carve((size_t)N_ * K_ * V_ * V_ * 2);
    bf16_t* Ar   = (bf16_t*)carve((size_t)N_ * K_ * V_ * V_ * 2);
    bf16_t* h    = (bf16_t*)carve((size_t)N_ * O2_ * V_ * 2);
    bf16_t* t2   = (bf16_t*)carve((size_t)N_ * OUT_ * K_ * V_ * 2);
    float*  y    = (float*)carve((size_t)N_ * OUT_ * V_ * 4);
    float*  scale = (float*)carve(O2_ * 4);
    float*  shift = (float*)carve(O2_ * 4);

    // 1. bf16 casts of x, conv_w
    long nx = (long)N_ * V_ * C_;
    cast_f32_to_bf16<<<(unsigned)((nx + 255) / 256), 256, 0, stream>>>(x, xb, nx);
    long nw = (long)O2_ * C_;
    cast_f32_to_bf16<<<(unsigned)((nw + 255) / 256), 256, 0, stream>>>(conv_w, wb, nw);

    // 2. A^T (f32 output) + bf16 operand copies
    prep_A<<<dim3(V_ / 32, V_ / 32, N_ * K_), dim3(32, 8), 0, stream>>>(A, At_out, Abt, Ar);

    // 3. GEMM1: h[n,o,v] = sum_c w[o,c]*x[n,v,c] + b[o]   (Bt = x[n] itself)
    gemm_bf16<<<dim3(O2_ / 256, V_ / 64, N_), 256, 0, stream>>>(
        wb, 0L, xb, (long)V_ * C_, conv_b, nullptr, h,
        O2_, C_, V_, 1, (long)O2_ * V_, 0L);

    // 4/5. BatchNorm stats + in-place normalize
    bn_stats<<<O2_, 256, 0, stream>>>(h, gamma, beta, scale, shift);
    long nh = (long)N_ * O2_ * V_;
    bn_apply<<<(unsigned)(nh / 256), 256, 0, stream>>>(h, scale, shift);

    // 6. GEMM2 per (n,k): t[c,e] = sum_m h[c,m]*A[m,e]; C -> t2[n,c,k*V+e]
    gemm_bf16<<<dim3(OUT_ / 256, V_ / 64, N_ * K_), 256, 0, stream>>>(
        h, (long)OUT_ * V_, Abt, (long)V_ * V_, nullptr, nullptr, t2,
        OUT_, V_, K_ * V_, K_, (long)OUT_ * K_ * V_, (long)V_);

    // 7. GEMM3 per n: y[c,m] = sum_{k,e} t2[c,k*V+e]*A[n,k,m,e]  (Bt = Ar)
    gemm_bf16<<<dim3(OUT_ / 256, V_ / 64, N_), 256, 0, stream>>>(
        t2, (long)OUT_ * K_ * V_, Ar, (long)V_ * K_ * V_, nullptr, y, nullptr,
        OUT_, K_ * V_, V_, 1, (long)OUT_ * V_, 0L);

    // 8. out = (1+eps)*x + y^T
    epilogue<<<dim3(V_ / 32, C_ / 32, N_), dim3(32, 8), 0, stream>>>(x, y, eps_p, out);
}
